// LGA_30571577213080
// MI455X (gfx1250) — compile-verified, matrix operand order checked by reading, not run on hardware
//
#include <hip/hip_runtime.h>
#include <hip/hip_bf16.h>

typedef __attribute__((ext_vector_type(16))) _Float16 v16h;
typedef __attribute__((ext_vector_type(8)))  _Float16 v8h;
typedef __attribute__((ext_vector_type(8)))  float    v8f;

namespace {
constexpr int   BATCH = 8192, NSQ = 16, DIM = 256, KPOS = 48;
constexpr int   NTOT = BATCH * NSQ;                 // 131072 rows for BN stats
constexpr float ATTN_SCALE = 0.17677669529663687f;  // 1/sqrt(32)
constexpr float EPSV = 1e-5f;
constexpr int   MB = 4, MROWS = MB * NSQ;           // 4 batches / workgroup -> M=64

// workspace byte offsets (16B aligned)
constexpr size_t WS_PSUM  = 0;        // 48 f32   : sum of pos rows
constexpr size_t WS_S     = 192;      // 48x48 f32: second moment of pos rows
constexpr size_t WS_SB    = 9472;     // 3 x (256 scale + 256 bias) f32 (folded BN)
constexpr size_t WS_W1H   = 15616;    // 3 x 256x64 f16 (K padded 48->64 with zeros)
constexpr size_t WS_W2H   = 113920;   // 3 x 256x256 f16
constexpr size_t WS_QKVH  = 507136;   // 768x256 f16
constexpr size_t WS_PROJH = 900352;   // 256x256 f16

// LDS byte offsets
constexpr int L_POS = 0;        // 64x64 f16 (K padded)
constexpr int L_XH  = 8192;     // 64x256 f16
constexpr int L_H   = 40960;    // 64x256 f16 hidden; later aliased as s_out
constexpr int L_Q   = 73728;    // 64x256 f16
constexpr int L_K   = 106496;   // 64x256 f16
constexpr int L_PQ  = 139264;   // 64x256 f16 pos_q
constexpr int L_PK  = 172032;   // 64x256 f16 pos_k
constexpr int L_VVT = 204800;   // [4][8][32][32] f16 : (v+pos_v) transposed, m padded
constexpr int L_AT  = 270336;   // 8 waves x 2 x 16x32 f16 : attn tiles, K padded
constexpr int LDS_BYTES = 286720;
} // namespace

// ---- WMMA fragment helpers (16-bit A/B layout) --------------
__device__ __forceinline__ v16h ldfrag(const _Float16* base, int ld, int lane) {
  // lane l: row = l&15 ; K chunks at c0=(l>>4)*8 and c0+16  (two b128 loads)
  const _Float16* p = base + (lane & 15) * ld + ((lane >> 4) << 3);
  v8h lo = *(const v8h*)(p);
  v8h hi = *(const v8h*)(p + 16);
  return __builtin_shufflevector(lo, hi, 0,1,2,3,4,5,6,7,8,9,10,11,12,13,14,15);
}
__device__ __forceinline__ v8f wmma32(v16h a, v16h b, v8f c) {
  return __builtin_amdgcn_wmma_f32_16x16x32_f16(false, a, false, b, (short)0, c,
                                                false, false);
}

// ---- Kernel 1: pos first/second moments (BN stats are linear in these) ------
__global__ __launch_bounds__(256) void bn_stats_reduce(
    const float* __restrict__ p_r, float* __restrict__ psum, float* __restrict__ S) {
  __shared__ float P[128][KPOS];
  const int tid = threadIdx.x;
  float accp[9];
#pragma unroll
  for (int i = 0; i < 9; ++i) accp[i] = 0.f;
  float accs = 0.f;
  for (int ch = blockIdx.x; ch < NTOT / 128; ch += gridDim.x) {
    __syncthreads();
    for (int i = tid; i < 128 * KPOS; i += 256)
      P[i / KPOS][i % KPOS] = p_r[(size_t)ch * 128 * KPOS + i];
    __syncthreads();
#pragma unroll
    for (int k = 0; k < 9; ++k) {
      int idx = tid + k * 256;
      if (idx < KPOS * KPOS) {
        int i = idx / KPOS, j = idx % KPOS;
        float a = 0.f;
        for (int r = 0; r < 128; ++r) a += P[r][i] * P[r][j];
        accp[k] += a;
      }
    }
    if (tid < KPOS) {
      float a = 0.f;
      for (int r = 0; r < 128; ++r) a += P[r][tid];
      accs += a;
    }
  }
#pragma unroll
  for (int k = 0; k < 9; ++k) {
    int idx = tid + k * 256;
    if (idx < KPOS * KPOS) atomicAdd(&S[idx], accp[k]);
  }
  if (tid < KPOS) atomicAdd(&psum[tid], accs);
}

// ---- Kernel 2: fold BN into per-channel scale/bias per branch ---------------
__global__ void bn_stats_finalize(
    const float* __restrict__ psum, const float* __restrict__ S,
    const float* __restrict__ qw1, const float* __restrict__ qb1,
    const float* __restrict__ qg,  const float* __restrict__ qbe,
    const float* __restrict__ kw1, const float* __restrict__ kb1,
    const float* __restrict__ kg,  const float* __restrict__ kbe,
    const float* __restrict__ vw1, const float* __restrict__ vb1,
    const float* __restrict__ vg,  const float* __restrict__ vbe,
    float* __restrict__ sb) {
  int t = blockIdx.x * blockDim.x + threadIdx.x;
  if (t >= 3 * DIM) return;
  int p = t >> 8, c = t & 255;
  const float* w1 = p == 0 ? qw1 : (p == 1 ? kw1 : vw1);
  const float* b1 = p == 0 ? qb1 : (p == 1 ? kb1 : vb1);
  const float* g  = p == 0 ? qg  : (p == 1 ? kg  : vg);
  const float* be = p == 0 ? qbe : (p == 1 ? kbe : vbe);
  float w[KPOS];
#pragma unroll
  for (int j = 0; j < KPOS; ++j) w[j] = w1[c * KPOS + j];
  const float invN = 1.0f / (float)NTOT;
  float m1 = 0.f;
  for (int j = 0; j < KPOS; ++j) m1 += w[j] * psum[j];
  m1 *= invN;
  float e2 = 0.f;
  for (int i = 0; i < KPOS; ++i) {
    float si = 0.f;
    for (int j = 0; j < KPOS; ++j) si += S[i * KPOS + j] * w[j];
    e2 += w[i] * si;
  }
  e2 *= invN;
  float b  = b1[c];
  float mu = m1 + b;                               // E[w.p + b]
  float var = e2 + 2.f * b * m1 + b * b - mu * mu; // E[h^2]-mu^2
  float sc = rsqrtf(var + EPSV) * g[c];
  sb[p * 512 + c]       = sc;                      // h_bn = acc*sc + bias
  sb[p * 512 + 256 + c] = (b - mu) * sc + be[c];
}

// ---- Kernel 3: convert all weights to f16 (row-major [out][in]) -------------
__global__ void weight_convert(
    const float* __restrict__ qkv_w, const float* __restrict__ proj_w,
    const float* __restrict__ qw1, const float* __restrict__ kw1,
    const float* __restrict__ vw1, const float* __restrict__ qw2,
    const float* __restrict__ kw2, const float* __restrict__ vw2,
    char* __restrict__ ws) {
  _Float16* w1h  = (_Float16*)(ws + WS_W1H);
  _Float16* w2h  = (_Float16*)(ws + WS_W2H);
  _Float16* qkvh = (_Float16*)(ws + WS_QKVH);
  _Float16* prjh = (_Float16*)(ws + WS_PROJH);
  const int n1 = 3 * 256 * 64, n2 = 3 * 256 * 256, nq = 768 * 256, np = 256 * 256;
  int stride = gridDim.x * blockDim.x;
  for (int i = blockIdx.x * blockDim.x + threadIdx.x; i < n1 + n2 + nq + np;
       i += stride) {
    if (i < n1) {
      int p = i >> 14, rem = i & 16383, r = rem >> 6, c = rem & 63;
      const float* w1 = p == 0 ? qw1 : (p == 1 ? kw1 : vw1);
      w1h[i] = (_Float16)(c < KPOS ? w1[r * KPOS + c] : 0.f);
    } else if (i < n1 + n2) {
      int j = i - n1, p = j >> 16;
      const float* w2 = p == 0 ? qw2 : (p == 1 ? kw2 : vw2);
      w2h[j] = (_Float16)w2[j & 65535];
    } else if (i < n1 + n2 + nq) {
      int j = i - n1 - n2;
      qkvh[j] = (_Float16)qkv_w[j];
    } else {
      int j = i - n1 - n2 - nq;
      prjh[j] = (_Float16)proj_w[j];
    }
  }
}

// ---- Kernel 4: fully fused pos-MLPs + QKV + attention + projection ----------
// Tile map: wave w owns M-tile (w>>1), N-tiles (w&1)+2i processed in PAIRS with
// two independent accumulator chains (fills WMMA->VALU hazard windows).
__global__ __launch_bounds__(256) void fused_attn(
    const float* __restrict__ x, const float* __restrict__ p_r,
    const float* __restrict__ proj_b, const float* __restrict__ qb2,
    const float* __restrict__ kb2, const float* __restrict__ vb2,
    const char* __restrict__ ws, float* __restrict__ out) {
  extern __shared__ char smem[];
  _Float16* s_pos  = (_Float16*)(smem + L_POS);
  _Float16* s_xh   = (_Float16*)(smem + L_XH);
  _Float16* s_h    = (_Float16*)(smem + L_H);
  _Float16* s_out  = s_h;  // s_h dead before attention writes s_out
  _Float16* s_q    = (_Float16*)(smem + L_Q);
  _Float16* s_k    = (_Float16*)(smem + L_K);
  _Float16* s_pq   = (_Float16*)(smem + L_PQ);
  _Float16* s_pk   = (_Float16*)(smem + L_PK);
  _Float16* s_vvT  = (_Float16*)(smem + L_VVT);
  _Float16* s_attn = (_Float16*)(smem + L_AT);

  const _Float16* w1h  = (const _Float16*)(ws + WS_W1H);
  const _Float16* w2h  = (const _Float16*)(ws + WS_W2H);
  const _Float16* qkvh = (const _Float16*)(ws + WS_QKVH);
  const _Float16* prjh = (const _Float16*)(ws + WS_PROJH);
  const float* sb = (const float*)(ws + WS_SB);

  const int tid = threadIdx.x;
  const int lane = tid & 31;
  const int wv = tid >> 5;            // 8 waves
  const int hi8 = (lane >> 4) << 3;   // C-layout row offset 0/8
  const int col = lane & 15;          // C-layout column
  const int b0 = blockIdx.x * MB;
  const int mt = wv >> 1;             // this wave's M-tile (batch within group)
  const int n0 = wv & 1;              // N-tile parity

  // init: zero vvT (+m-pad) and attn tiles (+K-pad), zero pos K-pad
  {
    _Float16* z = (_Float16*)(smem + L_VVT);
    for (int i = tid; i < (LDS_BYTES - L_VVT) / 2; i += 256) z[i] = (_Float16)0.f;
    for (int i = tid; i < MROWS * 64; i += 256) s_pos[i] = (_Float16)0.f;
  }
  __syncthreads();
  // stage pos (f32->f16, K padded to 64) and x (f32->f16)
  for (int i = tid; i < MROWS * KPOS; i += 256) {
    int r = i / KPOS, c = i - r * KPOS;
    s_pos[r * 64 + c] =
        (_Float16)p_r[(size_t)(b0 + (r >> 4)) * (NSQ * KPOS) + (r & 15) * KPOS + c];
  }
  for (int i = tid; i < MROWS * DIM; i += 256) {
    int r = i >> 8, c = i & 255;
    s_xh[i] = (_Float16)x[((size_t)(b0 + (r >> 4)) * NSQ + (r & 15)) * DIM + c];
  }
  __syncthreads();

  // ---- pos-MLP branches, order v, q, k (v lands transposed into vvT) ----
  for (int pb = 0; pb < 3; ++pb) {
    const int p = (pb == 0) ? 2 : (pb - 1);
    const _Float16* W1 = w1h + p * 256 * 64;
    const _Float16* W2 = w2h + p * 65536;
    const float* sc_p = sb + p * 512;
    const float* bs_p = sc_p + 256;
    const float* b2 = (p == 0) ? qb2 : (p == 1) ? kb2 : vb2;

    // GEMM1: (64x64pad) @ (64x256) + folded BN + relu -> s_h (f16)
    {
      v16h afr[2];
#pragma unroll
      for (int kt = 0; kt < 2; ++kt)
        afr[kt] = ldfrag(s_pos + mt * 16 * 64 + kt * 32, 64, lane);
      auto epi1 = [&](int nt, const v8f& acc) {
        int c = nt * 16 + col;
        float scv = sc_p[c], bsv = bs_p[c];
#pragma unroll
        for (int e = 0; e < 8; ++e) {
          float v = acc[e] * scv + bsv;
          v = v > 0.f ? v : 0.f;
          s_h[(mt * 16 + hi8 + e) * DIM + c] = (_Float16)v;
        }
      };
      for (int nt = n0; nt < 16; nt += 4) {
        v8f a0 = {}, a1 = {};
        const _Float16* B0 = W1 + nt * 16 * 64;
        const _Float16* B1 = W1 + (nt + 2) * 16 * 64;
#pragma unroll
        for (int kt = 0; kt < 2; ++kt) {
          a0 = wmma32(afr[kt], ldfrag(B0 + kt * 32, 64, lane), a0);
          a1 = wmma32(afr[kt], ldfrag(B1 + kt * 32, 64, lane), a1);
        }
        epi1(nt, a0);
        epi1(nt + 2, a1);
      }
    }
    __syncthreads();

    // GEMM2: (64x256) @ (256x256) -> pos_q / pos_k row-major, pos_v transposed
    {
      v16h afr[8];
#pragma unroll
      for (int kt = 0; kt < 8; ++kt)
        afr[kt] = ldfrag(s_h + mt * 16 * DIM + kt * 32, DIM, lane);
      auto epi2 = [&](int nt, const v8f& acc) {
        int c = nt * 16 + col;
        float b2v = b2[c];
        if (p == 2) {  // pos_v -> vvT[bt][h][d][m]
          int hh = c >> 5, d = c & 31;
#pragma unroll
          for (int e = 0; e < 8; ++e)
            s_vvT[(((mt * 8 + hh) * 32) + d) * 32 + hi8 + e] =
                (_Float16)(acc[e] + b2v);
        } else {
          _Float16* dst = (p == 0) ? s_pq : s_pk;
#pragma unroll
          for (int e = 0; e < 8; ++e)
            dst[(mt * 16 + hi8 + e) * DIM + c] = (_Float16)(acc[e] + b2v);
        }
      };
      for (int nt = n0; nt < 16; nt += 4) {
        v8f a0 = {}, a1 = {};
        const _Float16* B0 = W2 + nt * 16 * DIM;
        const _Float16* B1 = W2 + (nt + 2) * 16 * DIM;
#pragma unroll
        for (int kt = 0; kt < 8; ++kt) {
          a0 = wmma32(afr[kt], ldfrag(B0 + kt * 32, DIM, lane), a0);
          a1 = wmma32(afr[kt], ldfrag(B1 + kt * 32, DIM, lane), a1);
        }
        epi2(nt, a0);
        epi2(nt + 2, a1);
      }
    }
    __syncthreads();
  }

  // ---- QKV GEMM: (64x256) @ (256x768); v third accumulates into vvT ----
  {
    v16h afr[8];
#pragma unroll
    for (int kt = 0; kt < 8; ++kt)
      afr[kt] = ldfrag(s_xh + mt * 16 * DIM + kt * 32, DIM, lane);
    auto epiq = [&](int nt, const v8f& acc) {
      int c = nt * 16 + col;  // 0..767
      if (c < 256) {
#pragma unroll
        for (int e = 0; e < 8; ++e)
          s_q[(mt * 16 + hi8 + e) * DIM + c] = (_Float16)acc[e];
      } else if (c < 512) {
#pragma unroll
        for (int e = 0; e < 8; ++e)
          s_k[(mt * 16 + hi8 + e) * DIM + (c - 256)] = (_Float16)acc[e];
      } else {
        int cc = c - 512, hh = cc >> 5, d = cc & 31;
#pragma unroll
        for (int e = 0; e < 8; ++e) {
          int idx = (((mt * 8 + hh) * 32) + d) * 32 + hi8 + e;
          s_vvT[idx] = (_Float16)((float)s_vvT[idx] + acc[e]);  // v + pos_v
        }
      }
    };
    for (int nt = n0; nt < 48; nt += 4) {
      v8f a0 = {}, a1 = {};
      const _Float16* B0 = qkvh + nt * 16 * DIM;
      const _Float16* B1 = qkvh + (nt + 2) * 16 * DIM;
#pragma unroll
      for (int kt = 0; kt < 8; ++kt) {
        a0 = wmma32(afr[kt], ldfrag(B0 + kt * 32, DIM, lane), a0);
        a1 = wmma32(afr[kt], ldfrag(B1 + kt * 32, DIM, lane), a1);
      }
      epiq(nt, a0);
      epiq(nt + 2, a1);
    }
  }
  __syncthreads();

  // ---- attention: wave -> (batch, 4 heads), two heads in flight ----
  {
    const int bt = wv >> 1;
    const int h0 = (wv & 1) * 4;
    const _Float16* qB = s_q + bt * 16 * DIM;
    const _Float16* kB = s_k + bt * 16 * DIM;
    const _Float16* pqB = s_pq + bt * 16 * DIM;
    const _Float16* pkB = s_pk + bt * 16 * DIM;
    for (int hp = 0; hp < 2; ++hp) {
      const int ha = h0 + hp * 2, hb = ha + 1;
      const int ca = ha * 32, cb = hb * 32;
      v16h qfa = ldfrag(qB + ca, DIM, lane), qfb = ldfrag(qB + cb, DIM, lane);
      v16h kfa = ldfrag(kB + ca, DIM, lane), kfb = ldfrag(kB + cb, DIM, lane);
      v16h pqa = ldfrag(pqB + ca, DIM, lane), pqb = ldfrag(pqB + cb, DIM, lane);
      v16h pka = ldfrag(pkB + ca, DIM, lane), pkb = ldfrag(pkB + cb, DIM, lane);
      v8f da = {}, db = {};
      da = wmma32(qfa, kfa, da);  db = wmma32(qfb, kfb, db);   // q.k^T
      da = wmma32(qfa, pqa, da);  db = wmma32(qfb, pqb, db);   // q.pos_q^T
      da = wmma32(kfa, pka, da);  db = wmma32(kfb, pkb, db);   // k.pos_k^T
      float pra[8], prb[8];
#pragma unroll
      for (int e = 0; e < 8; ++e) {
        float xa = da[e] * ATTN_SCALE, xb = db[e] * ATTN_SCALE;
        float ma = xa, mb = xb;
#pragma unroll
        for (int msk = 1; msk < 16; msk <<= 1) {
          ma = fmaxf(ma, __shfl_xor(ma, msk, 32));
          mb = fmaxf(mb, __shfl_xor(mb, msk, 32));
        }
        float ea = __expf(xa - ma), eb = __expf(xb - mb);
        float sa = ea, sb2 = eb;
#pragma unroll
        for (int msk = 1; msk < 16; msk <<= 1) {
          sa += __shfl_xor(sa, msk, 32);
          sb2 += __shfl_xor(sb2, msk, 32);
        }
        pra[e] = ea / sa;
        prb[e] = eb / sb2;
      }
      _Float16* at0 = s_attn + (wv * 2) * 512;
      _Float16* at1 = at0 + 512;
#pragma unroll
      for (int e = 0; e < 8; ++e) {  // K-pad cols 16..31 stay 0
        at0[(hi8 + e) * 32 + col] = (_Float16)pra[e];
        at1[(hi8 + e) * 32 + col] = (_Float16)prb[e];
      }
      v16h afa = ldfrag(at0, 32, lane), afb = ldfrag(at1, 32, lane);
      const _Float16* vba = s_vvT + ((bt * 8 + ha) * 32) * 32;
      const _Float16* vbb = s_vvT + ((bt * 8 + hb) * 32) * 32;
#pragma unroll
      for (int dt = 0; dt < 2; ++dt) {
        v8f oa = wmma32(afa, ldfrag(vba + dt * 16 * 32, 32, lane), v8f{});
        v8f ob = wmma32(afb, ldfrag(vbb + dt * 16 * 32, 32, lane), v8f{});
#pragma unroll
        for (int e = 0; e < 8; ++e) {
          s_out[(bt * 16 + hi8 + e) * DIM + ca + dt * 16 + col] = (_Float16)oa[e];
          s_out[(bt * 16 + hi8 + e) * DIM + cb + dt * 16 + col] = (_Float16)ob[e];
        }
      }
    }
  }
  __syncthreads();

  // ---- projection: (64x256) @ (256x256) + bias -> global f32 ----
  {
    v16h afr[8];
#pragma unroll
    for (int kt = 0; kt < 8; ++kt)
      afr[kt] = ldfrag(s_out + mt * 16 * DIM + kt * 32, DIM, lane);
    auto epip = [&](int nt, const v8f& acc) {
      int c = nt * 16 + col;
      float bv = proj_b[c];
#pragma unroll
      for (int e = 0; e < 8; ++e)
        out[((size_t)(b0 + mt) * NSQ + hi8 + e) * DIM + c] = acc[e] + bv;
    };
    for (int nt = n0; nt < 16; nt += 4) {
      v8f a0 = {}, a1 = {};
      const _Float16* B0 = prjh + nt * 16 * DIM;
      const _Float16* B1 = prjh + (nt + 2) * 16 * DIM;
#pragma unroll
      for (int kt = 0; kt < 8; ++kt) {
        a0 = wmma32(afr[kt], ldfrag(B0 + kt * 32, DIM, lane), a0);
        a1 = wmma32(afr[kt], ldfrag(B1 + kt * 32, DIM, lane), a1);
      }
      epip(nt, a0);
      epip(nt + 2, a1);
    }
  }
}

extern "C" void kernel_launch(void* const* d_in, const int* in_sizes, int n_in,
                              void* d_out, int out_size, void* d_ws, size_t ws_size,
                              hipStream_t stream) {
  (void)in_sizes; (void)n_in; (void)out_size; (void)ws_size;
  const float* x      = (const float*)d_in[0];
  const float* p_r    = (const float*)d_in[1];
  const float* qkv_w  = (const float*)d_in[2];
  const float* proj_w = (const float*)d_in[3];
  const float* proj_b = (const float*)d_in[4];
  const float* qw1 = (const float*)d_in[5],  *qb1 = (const float*)d_in[6];
  const float* qg  = (const float*)d_in[7],  *qbe = (const float*)d_in[8];
  const float* qw2 = (const float*)d_in[9],  *qb2 = (const float*)d_in[10];
  const float* kw1 = (const float*)d_in[11], *kb1 = (const float*)d_in[12];
  const float* kg  = (const float*)d_in[13], *kbe = (const float*)d_in[14];
  const float* kw2 = (const float*)d_in[15], *kb2 = (const float*)d_in[16];
  const float* vw1 = (const float*)d_in[17], *vb1 = (const float*)d_in[18];
  const float* vg  = (const float*)d_in[19], *vbe = (const float*)d_in[20];
  const float* vw2 = (const float*)d_in[21], *vb2 = (const float*)d_in[22];
  char* ws = (char*)d_ws;

  hipMemsetAsync(ws, 0, WS_SB, stream);  // zero psum + S accumulators
  bn_stats_reduce<<<256, 256, 0, stream>>>(p_r, (float*)(ws + WS_PSUM),
                                           (float*)(ws + WS_S));
  bn_stats_finalize<<<3, 256, 0, stream>>>(
      (const float*)(ws + WS_PSUM), (const float*)(ws + WS_S),
      qw1, qb1, qg, qbe, kw1, kb1, kg, kbe, vw1, vb1, vg, vbe,
      (float*)(ws + WS_SB));
  weight_convert<<<512, 256, 0, stream>>>(qkv_w, proj_w, qw1, kw1, vw1, qw2, kw2,
                                          vw2, ws);
  hipFuncSetAttribute((const void*)fused_attn,
                      hipFuncAttributeMaxDynamicSharedMemorySize, LDS_BYTES);
  fused_attn<<<BATCH / MB, 256, LDS_BYTES, stream>>>(x, p_r, proj_b, qb2, kb2,
                                                     vb2, ws, (float*)d_out);
}
